// SelfAttention_83794811945113
// MI455X (gfx1250) — matrix operand order repaired and checked by reading, not compiled, not run on hardware
//
#include <hip/hip_runtime.h>
#include <hip/hip_bf16.h>

// ---------------------------------------------------------------------------
// Self-attention forward for MI455X (gfx1250), wave32, WMMA bf16 path.
// N=4, S=2048, E=1024, H=16, D=64.
// Round 3: fix async-to-LDS builtin pointer types (global/local int4 vectors).
// ---------------------------------------------------------------------------

typedef __bf16 bf16_t;
typedef __attribute__((ext_vector_type(16))) __bf16 v16bf;
typedef __attribute__((ext_vector_type(8)))  float  v8f;
typedef __attribute__((ext_vector_type(4)))  int    i4v;

#define EMB   1024
#define SLEN  2048
#define NBAT  4
#define NHEAD 16
#define HDIM  64

#if __has_builtin(__builtin_amdgcn_global_load_async_to_lds_b128)
#define ATTN_ASYNC 1
typedef __attribute__((address_space(1))) i4v gbl_i4v;
typedef __attribute__((address_space(3))) i4v lds_i4v;
#endif

__device__ static inline void wait_async0() {
#if __has_builtin(__builtin_amdgcn_s_wait_asynccnt)
  __builtin_amdgcn_s_wait_asynccnt(0);
#else
  asm volatile("s_wait_asynccnt 0x0" ::: "memory");
#endif
}

// ---- WMMA wrapper: D = A(16x32 bf16) * B(32x16 bf16) + C(16x16 f32) -------
__device__ static inline v8f wmma_bf16(v16bf a, v16bf b, v8f c) {
  return __builtin_amdgcn_wmma_f32_16x16x32_bf16(
      /*neg_a=*/false, a, /*neg_b=*/false, b,
      /*c_mod=*/(short)0, c, /*reuse_a=*/false, /*reuse_b=*/false);
}

union FragBits {
  uint4  u4[2];
  bf16_t h[16];
  v16bf  v;
};

// A-matrix 16x32 fragment from row-major source (element (r,k) at p[r*ld+k]).
// Per lane: row = lane&15, kbase = 8*(lane>>4); VGPR0-3 hold K=kb..kb+7,
// VGPR4-7 hold K=kb+16..kb+23  ->  two contiguous 16B runs.
__device__ static inline v16bf frag_a(const bf16_t* p, int ld, int lane) {
  const int r  = lane & 15;
  const int kb = (lane >> 4) << 3;
  const bf16_t* q = p + r * ld + kb;
  FragBits f;
  f.u4[0] = *reinterpret_cast<const uint4*>(q);
  f.u4[1] = *reinterpret_cast<const uint4*>(q + 16);
  return f.v;
}

// B-matrix 32x16 fragment from column-major source (element (k,c) at
// p[c*ld + k]). Per lane: col = lane&15, kbase = 8*(lane>>4);
// K = kb..kb+7 and kb+16..kb+23 -> two contiguous 16B runs.
__device__ static inline v16bf frag_bt(const bf16_t* p, int ld, int lane) {
  const int c  = lane & 15;
  const int kb = (lane >> 4) << 3;
  const bf16_t* q = p + c * ld + kb;
  FragBits f;
  f.u4[0] = *reinterpret_cast<const uint4*>(q);
  f.u4[1] = *reinterpret_cast<const uint4*>(q + 16);
  return f.v;
}

// ---------------------------------------------------------------------------
// Kernel 1: fused Q/K/V projection GEMM. C[8192,1024] = X[8192,1024] @ W.
// fp32 inputs converted to bf16 while staging into LDS; bf16 output.
// blockIdx.z: 0 = Q (scaled by 1/32), 1 = K, 2 = V.
// Block tile 128x128, 256 threads (8 waves), each wave 32x64, K-step 32.
// ---------------------------------------------------------------------------
__global__ __launch_bounds__(256) void qkv_proj_kernel(
    const float* __restrict__ Xq, const float* __restrict__ Xk,
    const float* __restrict__ Xv, const float* __restrict__ Wq,
    const float* __restrict__ Wk, const float* __restrict__ Wv,
    bf16_t* __restrict__ Qb, bf16_t* __restrict__ Kb, bf16_t* __restrict__ Vb) {
  __shared__ __align__(16) bf16_t As[128][40];  // [row][k], 80B row stride
  __shared__ __align__(16) bf16_t Bt[128][40];  // [col][k] (transposed W tile)

  const float* A;
  const float* W;
  bf16_t*      O;
  float        scale;
  if (blockIdx.z == 0)      { A = Xq; W = Wq; O = Qb; scale = 0.03125f; }
  else if (blockIdx.z == 1) { A = Xk; W = Wk; O = Kb; scale = 1.0f; }
  else                      { A = Xv; W = Wv; O = Vb; scale = 1.0f; }

  const int m0   = blockIdx.x * 128;
  const int n0   = blockIdx.y * 128;
  const int t    = threadIdx.x;
  const int lane = t & 31;
  const int w    = t >> 5;
  const int wr   = (w & 3) * 32;   // wave row offset in tile
  const int wc   = (w >> 2) * 64;  // wave col offset in tile

  v8f acc[2][4];
#pragma unroll
  for (int r = 0; r < 2; ++r)
#pragma unroll
    for (int j = 0; j < 4; ++j)
      acc[r][j] = {0.f, 0.f, 0.f, 0.f, 0.f, 0.f, 0.f, 0.f};

  for (int kt = 0; kt < EMB; kt += 32) {
    __syncthreads();
#pragma unroll
    for (int p = 0; p < 4; ++p) {
      const int idx = (p * 256 + t) * 4;
      {  // A tile: 128x32 fp32 -> bf16, packed 8B store
        const int r = idx >> 5, c = idx & 31;
        const float4 v =
            *reinterpret_cast<const float4*>(A + (size_t)(m0 + r) * EMB + kt + c);
        union { bf16_t h[4]; uint2 u; } pk;
        pk.h[0] = (bf16_t)v.x; pk.h[1] = (bf16_t)v.y;
        pk.h[2] = (bf16_t)v.z; pk.h[3] = (bf16_t)v.w;
        *reinterpret_cast<uint2*>(&As[r][c]) = pk.u;
      }
      {  // W tile: 32x128 fp32 -> bf16, stored transposed [col][k]
        const int r = idx >> 7, c = idx & 127;
        const float4 v =
            *reinterpret_cast<const float4*>(W + (size_t)(kt + r) * EMB + n0 + c);
        Bt[c][r]     = (bf16_t)v.x;
        Bt[c + 1][r] = (bf16_t)v.y;
        Bt[c + 2][r] = (bf16_t)v.z;
        Bt[c + 3][r] = (bf16_t)v.w;
      }
    }
    __syncthreads();

    const v16bf a0 = frag_a(&As[wr][0], 40, lane);
    const v16bf a1 = frag_a(&As[wr + 16][0], 40, lane);
#pragma unroll
    for (int j = 0; j < 4; ++j) {
      const v16bf b = frag_bt(&Bt[wc + 16 * j][0], 40, lane);
      acc[0][j] = wmma_bf16(a0, b, acc[0][j]);
      acc[1][j] = wmma_bf16(a1, b, acc[1][j]);
    }
  }

  const int rbase = (lane >> 4) * 8;
  const int cl    = lane & 15;
#pragma unroll
  for (int rt = 0; rt < 2; ++rt)
#pragma unroll
    for (int j = 0; j < 4; ++j)
#pragma unroll
      for (int i = 0; i < 8; ++i) {
        const int r = m0 + wr + rt * 16 + rbase + i;
        const int c = n0 + wc + 16 * j + cl;
        O[(size_t)r * EMB + c] = (bf16_t)(acc[rt][j][i] * scale);
      }
}

// ---------------------------------------------------------------------------
// Kernel 2: flash attention. One block per (128 q-rows, head, batch).
// 8 waves x 16 q-rows. K/V streamed in 64-key chunks through LDS.
// K chunk staged via async global->LDS (ASYNCcnt); V staged transposed.
// ---------------------------------------------------------------------------
__global__ __launch_bounds__(256) void attn_kernel(
    const bf16_t* __restrict__ Qb, const bf16_t* __restrict__ Kb,
    const bf16_t* __restrict__ Vb, bf16_t* __restrict__ Ob) {
  __shared__ __align__(16) bf16_t Ks[64][72];     // [key][d], 144B row stride
  __shared__ __align__(16) bf16_t Vt[64][72];     // [d][key] (transposed)
  __shared__ __align__(16) bf16_t Ps[8][16][72];  // per-wave P: [row][key]

  const int n    = blockIdx.z;
  const int h    = blockIdx.y;
  const int q0   = blockIdx.x * 128;
  const int t    = threadIdx.x;
  const int lane = t & 31;
  const int w    = t >> 5;
  const int s0   = q0 + w * 16;  // this wave's first query row

  // Q fragments (16x64 -> two 16x32 A-frags), 128-bit global loads.
  const bf16_t* qp = Qb + ((size_t)(n * SLEN + s0) * EMB + h * HDIM);
  const v16bf qf0 = frag_a(qp, EMB, lane);
  const v16bf qf1 = frag_a(qp + 32, EMB, lane);

  v8f oacc[4];
#pragma unroll
  for (int c = 0; c < 4; ++c) oacc[c] = {0.f, 0.f, 0.f, 0.f, 0.f, 0.f, 0.f, 0.f};
  float rmax[8], rsum[8];
#pragma unroll
  for (int i = 0; i < 8; ++i) { rmax[i] = -1.0e30f; rsum[i] = 0.0f; }

  bf16_t* Pw = &Ps[w][0][0];

  for (int kc = 0; kc < SLEN; kc += 64) {
    __syncthreads();

#ifdef ATTN_ASYNC
    // K chunk: 64x64 bf16 = 512 x 16B async transfers, 2 per thread.
#pragma unroll
    for (int q = 0; q < 2; ++q) {
      const int ci = t * 2 + q;       // 0..511
      const int r  = ci >> 3;         // key row
      const int ce = (ci & 7) * 8;    // element offset (8 bf16 = 16B)
      const bf16_t* gsrc =
          Kb + (size_t)(n * SLEN + kc + r) * EMB + h * HDIM + ce;
      __builtin_amdgcn_global_load_async_to_lds_b128(
          (gbl_i4v*)gsrc, (lds_i4v*)&Ks[r][ce], 0, 0);
    }
#else
#pragma unroll
    for (int p = 0; p < 4; ++p) {
      const int idx = (p * 256 + t) * 4;
      const int r = idx >> 6, c = idx & 63;
      *reinterpret_cast<uint2*>(&Ks[r][c]) = *reinterpret_cast<const uint2*>(
          Kb + (size_t)(n * SLEN + kc + r) * EMB + h * HDIM + c);
    }
#endif

    // V chunk staged transposed: Vt[d][key].
#pragma unroll
    for (int p = 0; p < 4; ++p) {
      const int idx = (p * 256 + t) * 4;
      const int r = idx >> 6, c = idx & 63;
      union { uint2 u; bf16_t h[4]; } vv;
      vv.u = *reinterpret_cast<const uint2*>(
          Vb + (size_t)(n * SLEN + kc + r) * EMB + h * HDIM + c);
      Vt[c][r]     = vv.h[0];
      Vt[c + 1][r] = vv.h[1];
      Vt[c + 2][r] = vv.h[2];
      Vt[c + 3][r] = vv.h[3];
    }

#ifdef ATTN_ASYNC
    wait_async0();
#endif
    __syncthreads();

    // Scores: S(16x64) = Q(16x64) x K^T(64x64); q already scaled by 1/32.
    v8f sacc[4];
#pragma unroll
    for (int j = 0; j < 4; ++j) sacc[j] = {0.f, 0.f, 0.f, 0.f, 0.f, 0.f, 0.f, 0.f};
#pragma unroll
    for (int kk = 0; kk < 2; ++kk) {
      const v16bf aq = kk ? qf1 : qf0;
#pragma unroll
      for (int j = 0; j < 4; ++j) {
        const v16bf bk = frag_bt(&Ks[16 * j][32 * kk], 72, lane);
        sacc[j] = wmma_bf16(aq, bk, sacc[j]);
      }
    }

    // Online softmax update, one row slot (accumulator VGPR) at a time.
#pragma unroll
    for (int i = 0; i < 8; ++i) {
      float m = fmaxf(fmaxf(sacc[0][i], sacc[1][i]),
                      fmaxf(sacc[2][i], sacc[3][i]));
#pragma unroll
      for (int d = 1; d < 16; d <<= 1) m = fmaxf(m, __shfl_xor(m, d, 32));
      const float mnew = fmaxf(rmax[i], m);
      const float corr = __expf(rmax[i] - mnew);
      rmax[i] = mnew;
      float s = 0.0f;
#pragma unroll
      for (int j = 0; j < 4; ++j) {
        const float pv = __expf(sacc[j][i] - mnew);
        sacc[j][i] = pv;
        s += pv;
      }
#pragma unroll
      for (int d = 1; d < 16; d <<= 1) s += __shfl_xor(s, d, 32);
      rsum[i] = rsum[i] * corr + s;
#pragma unroll
      for (int c = 0; c < 4; ++c) oacc[c][i] *= corr;
    }

    // C-layout -> A-layout for P via per-wave LDS tile (wave-local).
    {
      const int rr = (lane >> 4) * 8;
      const int cl = lane & 15;
#pragma unroll
      for (int j = 0; j < 4; ++j)
#pragma unroll
        for (int i = 0; i < 8; ++i)
          Pw[(rr + i) * 72 + 16 * j + cl] = (bf16_t)sacc[j][i];
    }

    // O += P(16x64) x V(64x64); V read from transposed tile (contiguous).
#pragma unroll
    for (int kk = 0; kk < 2; ++kk) {
      const v16bf ap = frag_a(Pw + 32 * kk, 72, lane);
#pragma unroll
      for (int c = 0; c < 4; ++c) {
        const v16bf bv = frag_bt(&Vt[16 * c][32 * kk], 72, lane);
        oacc[c] = wmma_bf16(ap, bv, oacc[c]);
      }
    }
  }

  // Normalize and store bf16 output at [n, s, h*64 + d].
  const int rbase = (lane >> 4) * 8;
  const int cl    = lane & 15;
#pragma unroll
  for (int c = 0; c < 4; ++c)
#pragma unroll
    for (int i = 0; i < 8; ++i) {
      const int srow = s0 + rbase + i;
      const int dcol = 16 * c + cl;
      const float o  = oacc[c][i] / rsum[i];
      Ob[(size_t)(n * SLEN + srow) * EMB + h * HDIM + dcol] = (bf16_t)o;
    }
}

// ---------------------------------------------------------------------------
// Kernel 3: output projection. out[8192,1024] = Attn(bf16) @ W_o + b_o (f32).
// ---------------------------------------------------------------------------
__global__ __launch_bounds__(256) void out_proj_kernel(
    const bf16_t* __restrict__ Ab, const float* __restrict__ W,
    const float* __restrict__ bias, float* __restrict__ out) {
  __shared__ __align__(16) bf16_t As[128][40];
  __shared__ __align__(16) bf16_t Bt[128][40];

  const int m0   = blockIdx.x * 128;
  const int n0   = blockIdx.y * 128;
  const int t    = threadIdx.x;
  const int lane = t & 31;
  const int w    = t >> 5;
  const int wr   = (w & 3) * 32;
  const int wc   = (w >> 2) * 64;

  v8f acc[2][4];
#pragma unroll
  for (int r = 0; r < 2; ++r)
#pragma unroll
    for (int j = 0; j < 4; ++j)
      acc[r][j] = {0.f, 0.f, 0.f, 0.f, 0.f, 0.f, 0.f, 0.f};

  for (int kt = 0; kt < EMB; kt += 32) {
    __syncthreads();
#pragma unroll
    for (int p = 0; p < 4; ++p) {
      const int idx = (p * 256 + t) * 4;
      {  // A tile already bf16: copy 4 elements as uint2
        const int r = idx >> 5, c = idx & 31;
        *reinterpret_cast<uint2*>(&As[r][c]) = *reinterpret_cast<const uint2*>(
            Ab + (size_t)(m0 + r) * EMB + kt + c);
      }
      {  // W tile fp32 -> bf16, stored transposed [col][k]
        const int r = idx >> 7, c = idx & 127;
        const float4 v =
            *reinterpret_cast<const float4*>(W + (size_t)(kt + r) * EMB + n0 + c);
        Bt[c][r]     = (bf16_t)v.x;
        Bt[c + 1][r] = (bf16_t)v.y;
        Bt[c + 2][r] = (bf16_t)v.z;
        Bt[c + 3][r] = (bf16_t)v.w;
      }
    }
    __syncthreads();

    const v16bf a0 = frag_a(&As[wr][0], 40, lane);
    const v16bf a1 = frag_a(&As[wr + 16][0], 40, lane);
#pragma unroll
    for (int j = 0; j < 4; ++j) {
      const v16bf b = frag_bt(&Bt[wc + 16 * j][0], 40, lane);
      acc[0][j] = wmma_bf16(a0, b, acc[0][j]);
      acc[1][j] = wmma_bf16(a1, b, acc[1][j]);
    }
  }

  const int rbase = (lane >> 4) * 8;
  const int cl    = lane & 15;
#pragma unroll
  for (int rt = 0; rt < 2; ++rt)
#pragma unroll
    for (int j = 0; j < 4; ++j)
#pragma unroll
      for (int i = 0; i < 8; ++i) {
        const int r = m0 + wr + rt * 16 + rbase + i;
        const int c = n0 + wc + 16 * j + cl;
        out[(size_t)r * EMB + c] = acc[rt][j][i] + bias[c];
      }
}

// ---------------------------------------------------------------------------
extern "C" void kernel_launch(void* const* d_in, const int* in_sizes, int n_in,
                              void* d_out, int out_size, void* d_ws,
                              size_t ws_size, hipStream_t stream) {
  const float* values  = (const float*)d_in[0];
  const float* keys    = (const float*)d_in[1];
  const float* queries = (const float*)d_in[2];
  const float* W_v     = (const float*)d_in[3];
  const float* W_k     = (const float*)d_in[4];
  const float* W_q     = (const float*)d_in[5];
  const float* W_o     = (const float*)d_in[6];
  const float* b_o     = (const float*)d_in[7];
  float*       out     = (float*)d_out;

  // Workspace layout: 4 bf16 buffers of N*S*E elements (16 MB each, 64 MB).
  const size_t elems = (size_t)NBAT * SLEN * EMB;
  bf16_t* qb = (bf16_t*)d_ws;
  bf16_t* kb = qb + elems;
  bf16_t* vb = kb + elems;
  bf16_t* ob = vb + elems;

  // 1) Q/K/V projections (z selects matrix; Q pre-scaled by 1/sqrt(E)=1/32).
  qkv_proj_kernel<<<dim3(64, 8, 3), 256, 0, stream>>>(
      queries, keys, values, W_q, W_k, W_v, qb, kb, vb);

  // 2) Flash attention: grid (S/128, H, N).
  attn_kernel<<<dim3(16, 16, 4), 256, 0, stream>>>(qb, kb, vb, ob);

  // 3) Output projection + bias.
  out_proj_kernel<<<dim3(64, 8, 1), 256, 0, stream>>>(ob, W_o, b_o, out);
}
